// LukeEntityAwareAttentionModel_17789754540297
// MI455X (gfx1250) — matrix operand order repaired and checked by reading, not compiled, not run on hardware
//
#include <hip/hip_runtime.h>
#include <math.h>

// ---------------- constants ----------------
#define B_ 8
#define SW_ 512
#define SE_ 128
#define S_ 640
#define D_ 1024
#define H_ 16
#define HD_ 64
#define FF_ 4096
#define SCALE_ 0.125f            // 1/sqrt(64)
#define EPS_ 1e-12f
#define WORD_TOTAL_ ((size_t)B_ * SW_ * D_)

typedef __bf16 bf16;
typedef __attribute__((ext_vector_type(16))) __bf16 v16bf;
typedef __attribute__((ext_vector_type(8)))  float  v8f;

// epilogue modes
#define EPI_QK       0
#define EPI_VT       1
#define EPI_BIAS_RES 2
#define EPI_GELU     3
#define EPI_CTX      4

// ---------------- TDM availability ----------------
#if defined(__has_builtin)
#if __has_builtin(__builtin_amdgcn_tensor_load_to_lds) && __has_builtin(__builtin_amdgcn_s_wait_tensorcnt)
#define USE_TDM 1
#endif
#endif
#ifndef USE_TDM
#define USE_TDM 0
#endif

#if USE_TDM
typedef unsigned int tdm_u32x4 __attribute__((ext_vector_type(4)));
typedef int          tdm_i32x4 __attribute__((ext_vector_type(4)));
typedef int          tdm_i32x8 __attribute__((ext_vector_type(8)));

// 2D tile load via Tensor Data Mover: tileRows rows of tileD0 bf16 each,
// row stride = strideElems (bf16 units), packed contiguously into LDS.
static __device__ __forceinline__ void tdm_load_2d(const bf16* g, unsigned ldsOff,
                                                   unsigned tileD0, unsigned tileRows,
                                                   unsigned long long strideElems) {
    unsigned long long ga = (unsigned long long)(uintptr_t)g;
    tdm_u32x4 g0;
    g0[0] = 1u;                                            // count=1, user mode
    g0[1] = ldsOff;                                        // lds_addr
    g0[2] = (unsigned)(ga & 0xFFFFFFFFu);                  // global_addr[31:0]
    g0[3] = (unsigned)((ga >> 32) & 0x01FFFFFFu) | 0x80000000u; // addr[56:32] | type=2

    tdm_i32x8 g1;
    g1[0] = (int)(1u << 16);                               // data_size=1 (2 bytes)
    g1[1] = (int)((tileD0 & 0xFFFFu) << 16);               // tensor_dim0[15:0]
    g1[2] = (int)((tileD0 >> 16) | ((tileRows & 0xFFFFu) << 16)); // dim0 hi | dim1 lo
    g1[3] = (int)((tileRows >> 16) | (tileD0 << 16));      // dim1 hi | tile_dim0
    g1[4] = (int)(tileRows & 0xFFFFu);                     // tile_dim1 (tile_dim2=0)
    g1[5] = (int)(unsigned)(strideElems & 0xFFFFFFFFu);    // tensor_dim0_stride lo
    g1[6] = (int)(unsigned)((strideElems >> 32) & 0xFFFFu);// stride hi (dim1_stride=0)
    g1[7] = 0;

    tdm_i32x4 gz = {0, 0, 0, 0};
#if defined(__clang_major__) && (__clang_major__ >= 23)
    tdm_i32x8 gz8 = {0, 0, 0, 0, 0, 0, 0, 0};
    __builtin_amdgcn_tensor_load_to_lds(g0, g1, gz, gz, gz8, 0);
#else
    __builtin_amdgcn_tensor_load_to_lds(g0, g1, gz, gz, 0);
#endif
}
#endif // USE_TDM

// ---------------- helpers ----------------
static __device__ __forceinline__ bf16 f2bf(float x) {
    unsigned u = __float_as_uint(x);
    unsigned r = (u + 0x7FFFu + ((u >> 16) & 1u)) >> 16;
    union { unsigned short s; bf16 b; } c;
    c.s = (unsigned short)r;
    return c.b;
}

union FragU { v16bf v; uint4 u[2]; };

// A fragment: two 16-byte chunks (K offsets half*8 and half*8+16)
static __device__ __forceinline__ v16bf ld_frag2(const bf16* p0, const bf16* p1) {
    FragU f;
    f.u[0] = *(const uint4*)p0;
    f.u[1] = *(const uint4*)p1;
    return f.v;
}
// contiguous 32-byte load (unused: B fragments use ld_frag2 like A)
static __device__ __forceinline__ v16bf ld_frag32(const bf16* p) {
    FragU f;
    const uint4* q = (const uint4*)p;
    f.u[0] = q[0];
    f.u[1] = q[1];
    return f.v;
}

static __device__ __forceinline__ v8f wmma_bf16(v16bf a, v16bf b, v8f c) {
    return __builtin_amdgcn_wmma_f32_16x16x32_bf16(false, a, false, b, (short)0, c, false, false);
}

// ---------------- pack activations ----------------
__global__ void pack_acts_kernel(const float* __restrict__ wh, const float* __restrict__ eh,
                                 bf16* __restrict__ whB, bf16* __restrict__ ehB,
                                 bf16* __restrict__ hidB, float* __restrict__ hidF) {
    size_t idx = (size_t)blockIdx.x * blockDim.x + threadIdx.x;
    size_t total = (size_t)B_ * S_ * D_;
    if (idx >= total) return;
    int d = (int)(idx % D_);
    int s = (int)((idx / D_) % S_);
    int b = (int)(idx / ((size_t)S_ * D_));
    float v;
    if (s < SW_) {
        v = wh[((size_t)b * SW_ + s) * D_ + d];
        whB[((size_t)b * SW_ + s) * D_ + d] = f2bf(v);
    } else {
        v = eh[((size_t)b * SE_ + (s - SW_)) * D_ + d];
        ehB[((size_t)b * SE_ + (s - SW_)) * D_ + d] = f2bf(v);
    }
    hidF[idx] = v;
    hidB[idx] = f2bf(v);
}

// ---------------- weight transpose+pack: Wt[n][k] = (bf16)W[k][n] ----------------
__global__ void transpose_pack_kernel(const float* __restrict__ W, bf16* __restrict__ Wt,
                                      int K, int N) {
    __shared__ float t[32][33];
    int n0 = blockIdx.x * 32;
    int k0 = blockIdx.y * 32;
    int tx = threadIdx.x;
    for (int j = threadIdx.y; j < 32; j += 8)
        t[j][tx] = W[(size_t)(k0 + j) * N + n0 + tx];
    __syncthreads();
    for (int j = threadIdx.y; j < 32; j += 8)
        Wt[(size_t)(n0 + j) * K + k0 + tx] = f2bf(t[tx][j]);
}

// ---- tiled WMMA GEMM: BM=128, BN=64, BK=64, 8 waves (4x2), 32x32/wave,
// ---- double-buffered TDM pipeline (DMA of tile i+1 overlaps WMMA of tile i)
__global__ void gemm_wmma_kernel(const bf16* __restrict__ A, const bf16* __restrict__ Bt, int K,
                                 unsigned long long aBatch, unsigned long long bBatch,
                                 int mode, const float* __restrict__ bias,
                                 const float* __restrict__ res,
                                 float* __restrict__ outF, bf16* __restrict__ outB,
                                 int rowsPerB, int sOff, int ldc) {
    __shared__ __align__(16) bf16 As[2][128 * 64];
    __shared__ __align__(16) bf16 Bs[2][64 * 64];

    const int tid  = threadIdx.x;
    const int w    = tid >> 5;
    const int lane = tid & 31;
    const int wr   = w >> 1;       // wave row (0..3) -> 32 rows each
    const int wc   = w & 1;        // wave col (0..1) -> 32 cols each
    const int bm   = blockIdx.x * 128;
    const int bn   = blockIdx.y * 64;
    const int batch = blockIdx.z;

    A  += (size_t)batch * aBatch;
    Bt += (size_t)batch * bBatch;

    v8f z = {0.f, 0.f, 0.f, 0.f, 0.f, 0.f, 0.f, 0.f};
    v8f acc[2][2];
    acc[0][0] = z; acc[0][1] = z; acc[1][0] = z; acc[1][1] = z;

    const int half8  = (lane < 16) ? 0 : 8;
    const int half16 = (lane < 16) ? 0 : 16;
    const int mlane  = lane & 15;
    const int nsteps = K >> 6;     // K/64

#if USE_TDM
    const unsigned asOff = (unsigned)(uintptr_t)&As[0][0];
    const unsigned bsOff = (unsigned)(uintptr_t)&Bs[0][0];
    // prologue: DMA tile 0 into buffer 0 (each wave moves its own slice)
    tdm_load_2d(&A[(size_t)(bm + w * 16) * K], asOff + (unsigned)w * 16u * 128u,
                64u, 16u, (unsigned long long)K);
    tdm_load_2d(&Bt[(size_t)(bn + w * 8) * K], bsOff + (unsigned)w * 8u * 128u,
                64u, 8u, (unsigned long long)K);
    for (int it = 0; it < nsteps; ++it) {
        const int cur = it & 1;
        __builtin_amdgcn_s_wait_tensorcnt(0);   // my in-flight DMA (buf[cur]) done
        __syncthreads();                        // everyone's DMA done; buf[nxt] reads (it-1) done
        if (it + 1 < nsteps) {
            const int nxt = cur ^ 1;
            int k1 = (it + 1) << 6;
            tdm_load_2d(&A[(size_t)(bm + w * 16) * K + k1],
                        asOff + (unsigned)nxt * 16384u + (unsigned)w * 16u * 128u,
                        64u, 16u, (unsigned long long)K);
            tdm_load_2d(&Bt[(size_t)(bn + w * 8) * K + k1],
                        bsOff + (unsigned)nxt * 8192u + (unsigned)w * 8u * 128u,
                        64u, 8u, (unsigned long long)K);
        }
#pragma unroll
        for (int kk = 0; kk < 2; ++kk) {
            v16bf av[2], bv[2];
#pragma unroll
            for (int ti = 0; ti < 2; ++ti) {
                const bf16* ap = &As[cur][(wr * 32 + ti * 16 + mlane) * 64 + kk * 32 + half8];
                av[ti] = ld_frag2(ap, ap + 16);
            }
#pragma unroll
            for (int tj = 0; tj < 2; ++tj) {
                const bf16* bp = &Bs[cur][(wc * 32 + tj * 16 + mlane) * 64 + kk * 32 + half8];
                bv[tj] = ld_frag2(bp, bp + 16);
            }
#pragma unroll
            for (int ti = 0; ti < 2; ++ti)
#pragma unroll
                for (int tj = 0; tj < 2; ++tj)
                    acc[ti][tj] = wmma_bf16(av[ti], bv[tj], acc[ti][tj]);
        }
    }
#else
    // manual staging fallback (single buffer, BK=64)
    for (int it = 0; it < nsteps; ++it) {
        int k0 = it << 6;
#pragma unroll
        for (int i = 0; i < 4; ++i) {           // A tile: 128 rows x 8 chunks
            int c = tid + 256 * i;
            *(uint4*)&As[0][(c >> 3) * 64 + (c & 7) * 8] =
                *(const uint4*)&A[(size_t)(bm + (c >> 3)) * K + k0 + (c & 7) * 8];
        }
#pragma unroll
        for (int i = 0; i < 2; ++i) {           // B tile: 64 rows x 8 chunks
            int c = tid + 256 * i;
            *(uint4*)&Bs[0][(c >> 3) * 64 + (c & 7) * 8] =
                *(const uint4*)&Bt[(size_t)(bn + (c >> 3)) * K + k0 + (c & 7) * 8];
        }
        __syncthreads();
#pragma unroll
        for (int kk = 0; kk < 2; ++kk) {
            v16bf av[2], bv[2];
#pragma unroll
            for (int ti = 0; ti < 2; ++ti) {
                const bf16* ap = &As[0][(wr * 32 + ti * 16 + mlane) * 64 + kk * 32 + half8];
                av[ti] = ld_frag2(ap, ap + 16);
            }
#pragma unroll
            for (int tj = 0; tj < 2; ++tj) {
                const bf16* bp = &Bs[0][(wc * 32 + tj * 16 + mlane) * 64 + kk * 32 + half8];
                bv[tj] = ld_frag2(bp, bp + 16);
            }
#pragma unroll
            for (int ti = 0; ti < 2; ++ti)
#pragma unroll
                for (int tj = 0; tj < 2; ++tj)
                    acc[ti][tj] = wmma_bf16(av[ti], bv[tj], acc[ti][tj]);
        }
        __syncthreads();
    }
#endif

    // ---- epilogue ----
#pragma unroll
    for (int ti = 0; ti < 2; ++ti) {
#pragma unroll
        for (int tj = 0; tj < 2; ++tj) {
#pragma unroll
            for (int r = 0; r < 8; ++r) {
                int m = bm + wr * 32 + ti * 16 + r + ((lane >= 16) ? 8 : 0);
                int n = bn + wc * 32 + tj * 16 + mlane;
                float v = acc[ti][tj][r];
                if (mode == EPI_QK) {
                    int b = m / rowsPerB, s = m % rowsPerB + sOff;
                    int h = n >> 6, d = n & 63;
                    outB[(((size_t)b * H_ + h) * S_ + s) * HD_ + d] = f2bf(v + bias[n]);
                } else if (mode == EPI_VT) {
                    int b = m / rowsPerB, s = m % rowsPerB;
                    int h = n >> 6, d = n & 63;
                    outB[(((size_t)b * H_ + h) * HD_ + d) * S_ + s] = f2bf(v + bias[n]);
                } else if (mode == EPI_BIAS_RES) {
                    size_t o = (size_t)m * ldc + n;
                    outF[o] = v + bias[n] + res[o];
                } else if (mode == EPI_GELU) {
                    float x = v + bias[n];
                    float g = 0.5f * x * (1.0f + erff(x * 0.70710678118654752f));
                    outB[(size_t)m * ldc + n] = f2bf(g);
                } else { // EPI_CTX
                    int bb = batch >> 4, hh = batch & 15;
                    outB[((size_t)bb * S_ + m) * D_ + hh * HD_ + n] = f2bf(v);
                }
            }
        }
    }
}

// ---------------- fused score + softmax (per block: one (b,h), 16 queries) ----------------
__global__ void score_softmax_kernel(const bf16* __restrict__ Qw, const bf16* __restrict__ Qe,
                                     const bf16* __restrict__ Kb,
                                     const float* __restrict__ wmask,
                                     const float* __restrict__ emask,
                                     bf16* __restrict__ probs) {
    __shared__ __align__(16) bf16 lqw[16 * 64];
    __shared__ __align__(16) bf16 lqe[16 * 64];
    __shared__ float sc[16][640];
    __shared__ float red[16][16];
    __shared__ float rstatM[16];
    __shared__ float rstatS[16];

    const int tid  = threadIdx.x;
    const int w    = tid >> 5;
    const int lane = tid & 31;
    const int bh   = blockIdx.y;            // b*H + h
    const int b    = bh / H_;
    const int q0   = blockIdx.x * 16;

    // stage Q tiles (16 x 64 bf16 each)
    {
        int row = tid >> 4;
        int c4  = (tid & 15) * 4;
        size_t go = ((size_t)bh * S_ + q0 + row) * HD_ + c4;
        *(uint2*)&lqw[row * 64 + c4] = *(const uint2*)&Qw[go];
        *(uint2*)&lqe[row * 64 + c4] = *(const uint2*)&Qe[go];
    }
    __syncthreads();

    const int half8  = (lane < 16) ? 0 : 8;
    const int half16 = (lane < 16) ? 0 : 16;
    const int mlane  = lane & 15;

    v8f z = {0.f, 0.f, 0.f, 0.f, 0.f, 0.f, 0.f, 0.f};
    v8f acc[5];
#pragma unroll
    for (int i = 0; i < 5; ++i) acc[i] = z;

#pragma unroll
    for (int kk = 0; kk < 2; ++kk) {
        const bf16* aw0 = &lqw[mlane * 64 + kk * 32 + half8];
        const bf16* ae0 = &lqe[mlane * 64 + kk * 32 + half8];
        v16bf aw = ld_frag2(aw0, aw0 + 16);
        v16bf ae = ld_frag2(ae0, ae0 + 16);
#pragma unroll
        for (int i = 0; i < 5; ++i) {
            int tc  = w + 8 * i;              // key tile 0..39
            int key = tc * 16 + mlane;
            const bf16* kp = Kb + ((size_t)bh * S_ + key) * HD_ + kk * 32 + half8;
            v16bf bv = ld_frag2(kp, kp + 16);
            acc[i] = wmma_bf16((tc < 32) ? aw : ae, bv, acc[i]);
        }
    }

    // spill scaled+masked scores to LDS
#pragma unroll
    for (int i = 0; i < 5; ++i) {
        int key = (w + 8 * i) * 16 + mlane;
        float mk = (key < SW_) ? wmask[b * SW_ + key] : emask[b * SE_ + key - SW_];
#pragma unroll
        for (int r = 0; r < 8; ++r) {
            int m = r + ((lane >= 16) ? 8 : 0);
            sc[m][key] = acc[i][r] * SCALE_ + mk;
        }
    }
    __syncthreads();

    // row softmax: 16 threads per row
    const int row = tid >> 4;
    const int l   = tid & 15;
    float mx = -1e30f;
    for (int k = l; k < S_; k += 16) mx = fmaxf(mx, sc[row][k]);
    red[row][l] = mx;
    __syncthreads();
    if (l == 0) {
        float m = red[row][0];
        for (int j = 1; j < 16; ++j) m = fmaxf(m, red[row][j]);
        rstatM[row] = m;
    }
    __syncthreads();
    float rm = rstatM[row];
    float sum = 0.f;
    for (int k = l; k < S_; k += 16) {
        float e = __expf(sc[row][k] - rm);
        sc[row][k] = e;
        sum += e;
    }
    red[row][l] = sum;
    __syncthreads();
    if (l == 0) {
        float s = 0.f;
        for (int j = 0; j < 16; ++j) s += red[row][j];
        rstatS[row] = s;
    }
    __syncthreads();
    float inv = 1.0f / rstatS[row];
    size_t base = ((size_t)bh * S_ + q0 + row) * S_;
    for (int k = l; k < S_; k += 16)
        probs[base + k] = f2bf(sc[row][k] * inv);
}

// ---------------- row LayerNorm (1024 cols, 256 threads/row) ----------------
__global__ void layernorm_kernel(const float* __restrict__ in, const float* __restrict__ g,
                                 const float* __restrict__ be, float* __restrict__ outF,
                                 bf16* __restrict__ outB, int mode) {
    __shared__ float s1[256];
    __shared__ float s2[256];
    __shared__ float stat[2];
    const int row = blockIdx.x;
    const int tid = threadIdx.x;
    float a = 0.f, q = 0.f;
    float x[4];
#pragma unroll
    for (int j = 0; j < 4; ++j) {
        x[j] = in[(size_t)row * D_ + tid + j * 256];
        a += x[j];
        q += x[j] * x[j];
    }
    s1[tid] = a; s2[tid] = q;
    __syncthreads();
    for (int st = 128; st > 0; st >>= 1) {
        if (tid < st) { s1[tid] += s1[tid + st]; s2[tid] += s2[tid + st]; }
        __syncthreads();
    }
    if (tid == 0) {
        float mean = s1[0] * (1.0f / D_);
        float var  = s2[0] * (1.0f / D_) - mean * mean;
        stat[0] = mean;
        stat[1] = rsqrtf(var + EPS_);
    }
    __syncthreads();
    float mean = stat[0], inv = stat[1];
#pragma unroll
    for (int j = 0; j < 4; ++j) {
        int d = tid + j * 256;
        float v = g[d] * (x[j] - mean) * inv + be[d];
        if (mode == 0) {
            outF[(size_t)row * D_ + d] = v;
            outB[(size_t)row * D_ + d] = f2bf(v);
        } else {
            int b = row / S_, s = row % S_;
            size_t o = (s < SW_) ? (((size_t)b * SW_ + s) * D_ + d)
                                 : (WORD_TOTAL_ + ((size_t)b * SE_ + (s - SW_)) * D_ + d);
            outF[o] = v;
        }
    }
}

// ---------------- host ----------------
extern "C" void kernel_launch(void* const* d_in, const int* in_sizes, int n_in,
                              void* d_out, int out_size, void* d_ws, size_t ws_size,
                              hipStream_t stream) {
    const float* wh    = (const float*)d_in[0];
    const float* eh    = (const float*)d_in[1];
    const float* wmask = (const float*)d_in[2];
    const float* emask = (const float*)d_in[3];
    const float* Wq    = (const float*)d_in[4];
    const float* bq    = (const float*)d_in[5];
    const float* Ww2e  = (const float*)d_in[6];
    const float* bw2e  = (const float*)d_in[7];
    const float* We2w  = (const float*)d_in[8];
    const float* be2w  = (const float*)d_in[9];
    const float* We2e  = (const float*)d_in[10];
    const float* be2e  = (const float*)d_in[11];
    const float* Wk    = (const float*)d_in[12];
    const float* bk    = (const float*)d_in[13];
    const float* Wv    = (const float*)d_in[14];
    const float* bv    = (const float*)d_in[15];
    const float* Wo    = (const float*)d_in[16];
    const float* bo    = (const float*)d_in[17];
    const float* ln1g  = (const float*)d_in[18];
    const float* ln1b  = (const float*)d_in[19];
    const float* Wi    = (const float*)d_in[20];
    const float* bi    = (const float*)d_in[21];
    const float* Wout  = (const float*)d_in[22];
    const float* bout  = (const float*)d_in[23];
    const float* ln2g  = (const float*)d_in[24];
    const float* ln2b  = (const float*)d_in[25];

    char* ws = (char*)d_ws;
    size_t off = 0;
    auto alloc = [&](size_t bytes) -> char* {
        char* p = ws + off;
        off = (off + bytes + 255) & ~(size_t)255;
        return p;
    };

    const size_t TOK = (size_t)B_ * S_;          // 5120
    bf16* whB   = (bf16*)alloc((size_t)B_ * SW_ * D_ * 2);
    bf16* ehB   = (bf16*)alloc((size_t)B_ * SE_ * D_ * 2);
    bf16* hidB  = (bf16*)alloc(TOK * D_ * 2);
    float* hidF = (float*)alloc(TOK * D_ * 4);
    bf16* WqT   = (bf16*)alloc((size_t)D_ * D_ * 2);
    bf16* Ww2eT = (bf16*)alloc((size_t)D_ * D_ * 2);
    bf16* We2wT = (bf16*)alloc((size_t)D_ * D_ * 2);
    bf16* We2eT = (bf16*)alloc((size_t)D_ * D_ * 2);
    bf16* WkT   = (bf16*)alloc((size_t)D_ * D_ * 2);
    bf16* WvT   = (bf16*)alloc((size_t)D_ * D_ * 2);
    bf16* WoT   = (bf16*)alloc((size_t)D_ * D_ * 2);
    bf16* WiT   = (bf16*)alloc((size_t)D_ * FF_ * 2);
    bf16* WoutT = (bf16*)alloc((size_t)D_ * FF_ * 2);
    bf16* Qw    = (bf16*)alloc(TOK * D_ * 2);    // [b,h,s,d]
    bf16* Qe    = (bf16*)alloc(TOK * D_ * 2);
    bf16* Kb    = (bf16*)alloc(TOK * D_ * 2);
    bf16* Vt    = (bf16*)alloc(TOK * D_ * 2);    // [b,h,d,s]
    bf16* probs = (bf16*)alloc((size_t)B_ * H_ * S_ * S_ * 2);
    bf16* ctxB  = (bf16*)alloc(TOK * D_ * 2);
    float* attnPre  = (float*)alloc(TOK * D_ * 4);
    float* attnOutF = (float*)alloc(TOK * D_ * 4);
    bf16* attnOutB  = (bf16*)alloc(TOK * D_ * 2);
    bf16* interB    = (bf16*)alloc(TOK * FF_ * 2);
    float* outPre   = (float*)alloc(TOK * D_ * 4);
    (void)ws_size; (void)in_sizes; (void)n_in; (void)out_size;

    // 1) pack activations
    {
        size_t total = TOK * D_;
        pack_acts_kernel<<<dim3((unsigned)((total + 255) / 256)), 256, 0, stream>>>(
            wh, eh, whB, ehB, hidB, hidF);
    }
    // 2) pack weights (transposed)
    auto tpack = [&](const float* W, bf16* Wt, int K, int N) {
        transpose_pack_kernel<<<dim3(N / 32, K / 32), dim3(32, 8), 0, stream>>>(W, Wt, K, N);
    };
    tpack(Wq, WqT, D_, D_);     tpack(Ww2e, Ww2eT, D_, D_);
    tpack(We2w, We2wT, D_, D_); tpack(We2e, We2eT, D_, D_);
    tpack(Wk, WkT, D_, D_);     tpack(Wv, WvT, D_, D_);
    tpack(Wo, WoT, D_, D_);
    tpack(Wi, WiT, D_, FF_);
    tpack(Wout, WoutT, FF_, D_);

    auto gemm = [&](const bf16* A, const bf16* Bt, int M, int N, int K,
                    unsigned long long aB, unsigned long long bB, int batches, int mode,
                    const float* bias, const float* res, float* oF, bf16* oB,
                    int rowsPerB, int sOff, int ldc) {
        gemm_wmma_kernel<<<dim3(M / 128, N / 64, batches), 256, 0, stream>>>(
            A, Bt, K, aB, bB, mode, bias, res, oF, oB, rowsPerB, sOff, ldc);
    };

    // 3) projections (packed into head layouts)
    gemm(whB,  WqT,   B_ * SW_, D_, D_, 0, 0, 1, EPI_QK, bq,   nullptr, nullptr, Qw, SW_, 0,   0);
    gemm(whB,  Ww2eT, B_ * SW_, D_, D_, 0, 0, 1, EPI_QK, bw2e, nullptr, nullptr, Qe, SW_, 0,   0);
    gemm(ehB,  We2wT, B_ * SE_, D_, D_, 0, 0, 1, EPI_QK, be2w, nullptr, nullptr, Qw, SE_, SW_, 0);
    gemm(ehB,  We2eT, B_ * SE_, D_, D_, 0, 0, 1, EPI_QK, be2e, nullptr, nullptr, Qe, SE_, SW_, 0);
    gemm(hidB, WkT,   (int)TOK, D_, D_, 0, 0, 1, EPI_QK, bk,   nullptr, nullptr, Kb, S_,  0,   0);
    gemm(hidB, WvT,   (int)TOK, D_, D_, 0, 0, 1, EPI_VT, bv,   nullptr, nullptr, Vt, S_,  0,   0);

    // 4) fused scores + softmax -> probs (bf16)
    score_softmax_kernel<<<dim3(S_ / 16, B_ * H_), 256, 0, stream>>>(
        Qw, Qe, Kb, wmask, emask, probs);

    // 5) ctx = probs @ V  (batched over B*H), packed back to [b,s,D]
    gemm(probs, Vt, S_, HD_, S_, (unsigned long long)S_ * S_, (unsigned long long)HD_ * S_,
         B_ * H_, EPI_CTX, nullptr, nullptr, nullptr, ctxB, 0, 0, 0);

    // 6) attn output proj + residual, then LN1
    gemm(ctxB, WoT, (int)TOK, D_, D_, 0, 0, 1, EPI_BIAS_RES, bo, hidF, attnPre, nullptr, 0, 0, D_);
    layernorm_kernel<<<(unsigned)TOK, 256, 0, stream>>>(attnPre, ln1g, ln1b, attnOutF, attnOutB, 0);

    // 7) FFN
    gemm(attnOutB, WiT, (int)TOK, FF_, D_, 0, 0, 1, EPI_GELU, bi, nullptr, nullptr, interB, 0, 0, FF_);
    gemm(interB, WoutT, (int)TOK, D_, FF_, 0, 0, 1, EPI_BIAS_RES, bout, attnOutF, outPre, nullptr, 0, 0, D_);

    // 8) LN2 -> split word/entity output
    layernorm_kernel<<<(unsigned)TOK, 256, 0, stream>>>(outPre, ln2g, ln2b, (float*)d_out, nullptr, 1);
}
